// Encoder_60842506715336
// MI455X (gfx1250) — compile-verified, matrix operand order checked by reading, not run on hardware
//
#include <hip/hip_runtime.h>
#include <hip/hip_bf16.h>
#include <cstdint>
#include <cstddef>

typedef __attribute__((ext_vector_type(16))) _Float16 v16h;
typedef __attribute__((ext_vector_type(8)))  float    v8f;

#define DEVI __device__ __forceinline__

// ---------------------------------------------------------------------------
// Fragment loaders (CDNA5 16-bit A/B layout, CDNA5 ISA §7.12.2).
// Lane group g (lane>>4) owns two CONTIGUOUS 16-byte chunks of the 32-element
// K-span: halves [8g, 8g+8) and [16+8g, 16+8g+8). So a fragment is exactly
// two b128 loads at byte offsets 16g and 32+16g -- no per-lane permutes.
// ---------------------------------------------------------------------------
union HFrag { v16h v; uint4 q[2]; };

DEVI v16h load_frag_h(const _Float16* __restrict__ p, int g) {
  const uint4* q = (const uint4*)p;
  HFrag f;
  f.q[0] = q[g];
  f.q[1] = q[2 + g];
  return f.v;
}
DEVI v16h load_frag_h_pred(const _Float16* __restrict__ p, int g, bool ok) {
  const uint4* q = (const uint4*)p;
  uint4 z = make_uint4(0u, 0u, 0u, 0u);
  uint4 t0 = q[g];
  uint4 t1 = q[2 + g];
  HFrag f;
  f.q[0] = ok ? t0 : z;
  f.q[1] = ok ? t1 : z;
  return f.v;
}
DEVI v16h load_frag_f(const float* __restrict__ p, int g) {
  const float4* q = (const float4*)p;
  float4 a0 = q[2*g], a1 = q[2*g + 1], b0 = q[4 + 2*g], b1 = q[5 + 2*g];
  v16h f;
  f[0]  = (_Float16)a0.x; f[1]  = (_Float16)a0.y;
  f[2]  = (_Float16)a0.z; f[3]  = (_Float16)a0.w;
  f[4]  = (_Float16)a1.x; f[5]  = (_Float16)a1.y;
  f[6]  = (_Float16)a1.z; f[7]  = (_Float16)a1.w;
  f[8]  = (_Float16)b0.x; f[9]  = (_Float16)b0.y;
  f[10] = (_Float16)b0.z; f[11] = (_Float16)b0.w;
  f[12] = (_Float16)b1.x; f[13] = (_Float16)b1.y;
  f[14] = (_Float16)b1.z; f[15] = (_Float16)b1.w;
  return f;
}
DEVI v16h load_frag_any(const _Float16* __restrict__ p, int g) { return load_frag_h(p, g); }
DEVI v16h load_frag_any(const float* __restrict__ p, int g) { return load_frag_f(p, g); }

DEVI float gelu_tanh(float x) {
  float u = 0.7978845608028654f * (x + 0.044715f * x * x * x);
  return 0.5f * x * (1.0f + tanhf(u));
}
DEVI float wave_sum(float v) {
#pragma unroll
  for (int o = 16; o > 0; o >>= 1) v += __shfl_xor(v, o, 32);
  return v;
}

// ---------------------------------------------------------------------------
// Implicit-GEMM 3x3 conv, SAME pad. act: [B*H*W, Cin] NHWC f16 (Cin%32==0),
// wt: [9, Cout, Cin] f16, out: [B*H*W, Cout] fp32. One wave per 16x16 tile.
// Wave-uniform interior/boundary split: interior tiles (~86%+) run with zero
// predication selects; only halo tiles pay the 8-dword cndmask per fragment.
// ---------------------------------------------------------------------------
__global__ __launch_bounds__(256) void conv3x3_wmma_k(
    const _Float16* __restrict__ act, const _Float16* __restrict__ wt,
    float* __restrict__ out, int B, int H, int W, int Cin, int Cout)
{
  int wave = (int)((blockIdx.x * blockDim.x + threadIdx.x) >> 5);
  int lane = threadIdx.x & 31;
  int tilesN = Cout >> 4;
  int tilesM = (B * H * W) >> 4;
  if (wave >= tilesM * tilesN) return;
  int tm = wave / tilesN, tn = wave - tm * tilesN;
  int g = lane >> 4, mn = lane & 15;
  int p0 = tm << 4;
  int hw = H * W;
  int b = p0 / hw;
  int rem = p0 - b * hw;
  int y = rem / W;
  int x0 = rem - y * W;          // W%16==0 -> one tile = 16 consecutive x
  int col = (tn << 4) + mn;
  v8f acc = {};
  bool interior = (y >= 1) && (y + 1 < H) && (x0 >= 1) && (x0 + 16 < W);
  if (interior) {
#pragma unroll
    for (int dy = -1; dy <= 1; ++dy) {
#pragma unroll
      for (int dx = -1; dx <= 1; ++dx) {
        const _Float16* arow =
            act + ((size_t)((b * H + y + dy) * W + x0 + mn + dx)) * Cin;
        const _Float16* wrow =
            wt + ((size_t)(((dy+1)*3 + (dx+1)) * Cout + col)) * Cin;
        for (int k0 = 0; k0 < Cin; k0 += 32) {
          v16h av = load_frag_h(arow + k0, g);
          v16h bv = load_frag_h(wrow + k0, g);
          acc = __builtin_amdgcn_wmma_f32_16x16x32_f16(false, av, false, bv,
                                                       (short)0, acc, false, false);
        }
      }
    }
  } else {
#pragma unroll
    for (int dy = -1; dy <= 1; ++dy) {
      int yy = y + dy;
      bool yok = (yy >= 0) && (yy < H);
      int yc = yok ? yy : 0;
#pragma unroll
      for (int dx = -1; dx <= 1; ++dx) {
        int xx = x0 + mn + dx;
        bool ok = yok && (xx >= 0) && (xx < W);
        int xc = ok ? xx : 0;
        const _Float16* arow = act + ((size_t)((b * H + yc) * W + xc)) * Cin;
        const _Float16* wrow = wt + ((size_t)(((dy+1)*3 + (dx+1)) * Cout + col)) * Cin;
        for (int k0 = 0; k0 < Cin; k0 += 32) {
          v16h av = load_frag_h_pred(arow + k0, g, ok);
          v16h bv = load_frag_h(wrow + k0, g);
          acc = __builtin_amdgcn_wmma_f32_16x16x32_f16(false, av, false, bv,
                                                       (short)0, acc, false, false);
        }
      }
    }
  }
  float* obase = out + (size_t)p0 * Cout + col;
#pragma unroll
  for (int r = 0; r < 8; ++r) obase[(size_t)(g * 8 + r) * Cout] = acc[r];
}

// ---------------------------------------------------------------------------
// Generic GEMM: out[M,N] = act(A[M,K] * Wt[N,K]^T + b0 + b1)
// AT = float or _Float16 rows (converted to f16). actmode: 0 none,1 relu,2 gelu
// Output row remap: dst = bidx*outBStride + outBase + rr*N (+col)
// ---------------------------------------------------------------------------
template <typename AT>
__global__ __launch_bounds__(256) void gemm_wmma_k(
    const AT* __restrict__ A, const _Float16* __restrict__ Wt,
    const float* __restrict__ b0, const float* __restrict__ b1,
    float* __restrict__ out, int M, int K, int N,
    int mPerB, long outBStride, long outBase, int actmode)
{
  int wave = (int)((blockIdx.x * blockDim.x + threadIdx.x) >> 5);
  int lane = threadIdx.x & 31;
  int tilesN = N >> 4;
  long tiles = (long)(M >> 4) * tilesN;
  if (wave >= tiles) return;
  int tm = wave / tilesN, tn = wave - tm * tilesN;
  int g = lane >> 4, mn = lane & 15;
  int col = (tn << 4) + mn;
  const AT* arow = A + (size_t)(tm * 16 + mn) * K;
  const _Float16* wrow = Wt + (size_t)col * K;
  v8f acc = {};
  for (int k0 = 0; k0 < K; k0 += 32) {
    v16h av = load_frag_any(arow + k0, g);
    v16h bv = load_frag_h(wrow + k0, g);
    acc = __builtin_amdgcn_wmma_f32_16x16x32_f16(false, av, false, bv,
                                                 (short)0, acc, false, false);
  }
  float bb = (b0 ? b0[col] : 0.f) + (b1 ? b1[col] : 0.f);
#pragma unroll
  for (int r = 0; r < 8; ++r) {
    int mrow = (tm << 4) + g * 8 + r;
    float v = acc[r] + bb;
    if (actmode == 1) v = fmaxf(v, 0.f);
    else if (actmode == 2) v = gelu_tanh(v);
    int bidx = mrow / mPerB;
    int rr = mrow - bidx * mPerB;
    out[(long)bidx * outBStride + outBase + (long)rr * N + col] = v;
  }
}

// --------------------------- elementwise / reshape -------------------------
__global__ void nchw_pad_k(const float* __restrict__ x, _Float16* __restrict__ o,
                           int B, int C, int HW, int Cp)
{
  long tid = (long)blockIdx.x * blockDim.x + threadIdx.x;
  long tot = (long)B * HW * Cp;
  if (tid >= tot) return;
  int c = (int)(tid % Cp);
  long t = tid / Cp;
  int s = (int)(t % HW);
  int b = (int)(t / HW);
  float v = (c < C) ? x[((long)b * C + c) * HW + s] : 0.f;
  o[tid] = (_Float16)v;
}

__global__ void wt_conv3_k(const float* __restrict__ w, _Float16* __restrict__ o,
                           int O, int I, int Ip)
{
  long tid = (long)blockIdx.x * blockDim.x + threadIdx.x;
  long tot = 9L * O * Ip;
  if (tid >= tot) return;
  int i = (int)(tid % Ip);
  long t = tid / Ip;
  int oo = (int)(t % O);
  int k = (int)(t / O);
  o[tid] = (_Float16)((i < I) ? w[((long)oo * I + i) * 9 + k] : 0.f);
}

__global__ void wt_oi_k(const float* __restrict__ w, _Float16* __restrict__ o, long n)
{
  long tid = (long)blockIdx.x * blockDim.x + threadIdx.x;
  if (tid < n) o[tid] = (_Float16)w[tid];
}

__global__ void wt_lin_k(const float* __restrict__ w, _Float16* __restrict__ o,
                         int K, int N)   // w[K][N] -> o[N][K]
{
  long tid = (long)blockIdx.x * blockDim.x + threadIdx.x;
  long tot = (long)K * N;
  if (tid >= tot) return;
  int k = (int)(tid % K);
  int n = (int)(tid / K);
  o[tid] = (_Float16)w[(long)k * N + n];
}

#define BNCH 64
__global__ void bn_partial_k(const float* __restrict__ x, float* __restrict__ part,
                             int P, int C)
{
  int chunk = blockIdx.x;
  long r0 = (long)P * chunk / BNCH, r1 = (long)P * (chunk + 1) / BNCH;
  for (int c = threadIdx.x; c < C; c += blockDim.x) {
    float s = 0.f, s2 = 0.f;
    for (long r = r0; r < r1; ++r) { float v = x[r * C + c]; s += v; s2 += v * v; }
    part[((long)chunk * C + c) * 2 + 0] = s;
    part[((long)chunk * C + c) * 2 + 1] = s2;
  }
}
__global__ void bn_finalize_k(const float* __restrict__ part, float* __restrict__ st,
                              int P, int C)
{
  int c = blockIdx.x * blockDim.x + threadIdx.x;
  if (c >= C) return;
  float s = 0.f, s2 = 0.f;
  for (int k = 0; k < BNCH; ++k) {
    s += part[((long)k * C + c) * 2 + 0];
    s2 += part[((long)k * C + c) * 2 + 1];
  }
  float m = s / (float)P;
  float var = s2 / (float)P - m * m;
  st[c * 2 + 0] = m;
  st[c * 2 + 1] = rsqrtf(var + 1e-5f);
}
__global__ void bn_apply_k(const float* __restrict__ x, const float* __restrict__ st,
                           _Float16* __restrict__ oh, float* __restrict__ on,
                           int P, int C, int HW)
{
  long tid = (long)blockIdx.x * blockDim.x + threadIdx.x;
  long tot = (long)P * C;
  if (tid >= tot) return;
  int c = (int)(tid % C);
  long p = tid / C;
  float v = (x[tid] - st[c * 2]) * st[c * 2 + 1];
  v = fmaxf(v, 0.f);
  if (oh) oh[tid] = (_Float16)v;
  if (on) {
    int b = (int)(p / HW);
    int s = (int)(p - (long)b * HW);
    on[((long)b * C + c) * HW + s] = v;
  }
}

__global__ void maxpool2_k(const _Float16* __restrict__ in, _Float16* __restrict__ out,
                           int B, int H, int W, int C)
{
  long tid = (long)blockIdx.x * blockDim.x + threadIdx.x;
  int Ho = H >> 1, Wo = W >> 1;
  long tot = (long)B * Ho * Wo * C;
  if (tid >= tot) return;
  int c = (int)(tid % C);
  long t = tid / C;
  int xo = (int)(t % Wo); t /= Wo;
  int yo = (int)(t % Ho);
  int b = (int)(t / Ho);
  const _Float16* p = in + (((size_t)(b * H + 2 * yo) * W + 2 * xo) * C + c);
  float m = (float)p[0];
  m = fmaxf(m, (float)p[C]);
  m = fmaxf(m, (float)p[(size_t)W * C]);
  m = fmaxf(m, (float)p[(size_t)W * C + C]);
  out[tid] = (_Float16)m;
}

__global__ void softmax_k(float* __restrict__ a, long rows, int n)
{
  long r = (long)blockIdx.x * blockDim.x + threadIdx.x;
  if (r >= rows) return;
  float* p = a + r * n;
  float mx = -3.0e38f;
  for (int i = 0; i < n; ++i) mx = fmaxf(mx, p[i]);
  float s = 0.f;
  for (int i = 0; i < n; ++i) { float e = expf(p[i] - mx); p[i] = e; s += e; }
  float inv = 1.f / s;
  for (int i = 0; i < n; ++i) p[i] *= inv;
}

__global__ void add_ln_k(float* __restrict__ s, const float* __restrict__ a,
                         const float* __restrict__ g, const float* __restrict__ b,
                         int ntok)
{
  int tok = blockIdx.x;
  if (tok >= ntok) return;
  int lane = threadIdx.x;
  float v[4];
  float sum = 0.f;
#pragma unroll
  for (int i = 0; i < 4; ++i) {
    int c = lane + 32 * i;
    float x = s[(size_t)tok * 128 + c] + a[(size_t)tok * 128 + c];
    v[i] = x; sum += x;
  }
  sum = wave_sum(sum);
  float mean = sum * (1.f / 128.f);
  float s2 = 0.f;
#pragma unroll
  for (int i = 0; i < 4; ++i) { float d = v[i] - mean; s2 += d * d; }
  s2 = wave_sum(s2);
  float r = rsqrtf(s2 * (1.f / 128.f) + 1e-5f);
#pragma unroll
  for (int i = 0; i < 4; ++i) {
    int c = lane + 32 * i;
    s[(size_t)tok * 128 + c] = (v[i] - mean) * r * g[c] + b[c];
  }
}

// MS-deformable-attention sampling: one thread per (b,q,head,channel)
__global__ void msda_sample_k(const float* __restrict__ val, const float* __restrict__ off,
                              const float* __restrict__ aw, float* __restrict__ out,
                              int Lq, int L,
                              int h0, int w0, int s0, int h1, int w1, int s1,
                              int h2, int w2, int s2, int h3, int w3, int s3)
{
  long tid = (long)blockIdx.x * blockDim.x + threadIdx.x;
  long tot = (long)2 * Lq * 128;
  if (tid >= tot) return;
  int d = (int)(tid & 15);
  int hd = (int)((tid >> 4) & 7);
  long t = tid >> 7;
  int q = (int)(t % Lq);
  int b = (int)(t / Lq);
  int lh[4] = {h0, h1, h2, h3}, lw[4] = {w0, w1, w2, w3}, ls[4] = {s0, s1, s2, s3};
  int srcl = 0;
  for (int l = 1; l < L; ++l) if (q >= ls[l]) srcl = l;
  int sq = q - ls[srcl];
  float rx = ((float)(sq % lw[srcl]) + 0.5f) / (float)lw[srcl];
  float ry = ((float)(sq / lw[srcl]) + 0.5f) / (float)lh[srcl];
  long row = (long)b * Lq + q;
  const float* offr = off + row * (long)(64 * L);
  const float* awr = aw + row * (long)(32 * L);
  float acc = 0.f;
  for (int l = 0; l < L; ++l) {
    float wl = (float)lw[l], hl = (float)lh[l];
    int Wl = lw[l], Hl = lh[l];
    const float* vb = val + ((long)b * Lq + ls[l]) * 128 + hd * 16 + d;
    for (int p = 0; p < 4; ++p) {
      float ox = offr[((hd * L + l) * 4 + p) * 2 + 0];
      float oy = offr[((hd * L + l) * 4 + p) * 2 + 1];
      float a = awr[hd * (4 * L) + l * 4 + p];
      float xf = (rx + ox / wl) * wl - 0.5f;
      float yf = (ry + oy / hl) * hl - 0.5f;
      float x0f = floorf(xf), y0f = floorf(yf);
      float wx = xf - x0f, wy = yf - y0f;
      int x0i = (int)x0f, y0i = (int)y0f;
      float v00 = 0.f, v01 = 0.f, v10 = 0.f, v11 = 0.f;
      if (y0i >= 0 && y0i < Hl) {
        if (x0i >= 0 && x0i < Wl)         v00 = vb[(long)(y0i * Wl + x0i) * 128];
        if (x0i + 1 >= 0 && x0i + 1 < Wl) v01 = vb[(long)(y0i * Wl + x0i + 1) * 128];
      }
      if (y0i + 1 >= 0 && y0i + 1 < Hl) {
        if (x0i >= 0 && x0i < Wl)         v10 = vb[(long)((y0i + 1) * Wl + x0i) * 128];
        if (x0i + 1 >= 0 && x0i + 1 < Wl) v11 = vb[(long)((y0i + 1) * Wl + x0i + 1) * 128];
      }
      acc += a * (v00 * (1 - wx) * (1 - wy) + v01 * wx * (1 - wy) +
                  v10 * (1 - wx) * wy + v11 * wx * wy);
    }
  }
  out[row * 128 + hd * 16 + d] = acc;
}

__global__ void tok_to_f16_k(const float* __restrict__ src, _Float16* __restrict__ dst,
                             int Lq, int start, int hw)
{
  long tid = (long)blockIdx.x * blockDim.x + threadIdx.x;
  long tot = (long)2 * hw * 128;
  if (tid >= tot) return;
  int c = (int)(tid % 128);
  long t = tid / 128;
  int s = (int)(t % hw);
  int b = (int)(t / hw);
  dst[tid] = (_Float16)src[((long)b * Lq + start + s) * 128 + c];
}

// ===========================================================================
extern "C" void kernel_launch(void* const* d_in, const int* in_sizes, int n_in,
                              void* d_out, int out_size, void* d_ws, size_t ws_size,
                              hipStream_t stream)
{
  (void)in_sizes; (void)n_in; (void)out_size; (void)ws_size;
  auto F = [&](int i) { return (const float*)d_in[i]; };

  // --- output slices (fp32 NCHW, concatenated) ---
  float* out = (float*)d_out;
  float* ox1 = out;
  float* ox2 = out + 8388608;
  float* ox3 = out + 12582912;
  float* ox4 = out + 14680064;
  float* oxb = out + 15728640;

  // --- bump allocator over d_ws ---
  char* wsp = (char*)d_ws;
  size_t wof = 0;
  auto balloc = [&](size_t bytes) -> void* {
    size_t a = (wof + 255) & ~(size_t)255;
    wof = a + bytes;
    return (void*)(wsp + a);
  };
  auto ah = [&](size_t n) { return (_Float16*)balloc(n * sizeof(_Float16)); };
  auto af = [&](size_t n) { return (float*)balloc(n * sizeof(float)); };

  // --- persistent f16 NHWC activations ---
  _Float16* x2f = ah((size_t)2 * 16384 * 128);
  _Float16* x3f = ah((size_t)2 * 4096 * 256);
  _Float16* x4f = ah((size_t)2 * 1024 * 512);
  _Float16* x5f = ah((size_t)2 * 256 * 1024);
  _Float16* t1f = ah((size_t)2 * 16384 * 128);
  _Float16* t2f = ah((size_t)2 * 4096 * 128);
  _Float16* t3f = ah((size_t)2 * 1024 * 128);
  _Float16* t4f = ah((size_t)2 * 256 * 128);
  size_t ck = wof;

  // --- conv-stage scratch ---
  _Float16* wtbuf = ah((size_t)9 * 1024 * 1024);
  float* gbuf = af((size_t)131072 * 64);
  float* bnpart = af((size_t)BNCH * 1024 * 2);
  float* bnstats = af((size_t)1024 * 2);
  _Float16* pa = ah((size_t)131072 * 64);
  _Float16* pb = ah((size_t)131072 * 64);

  auto gb = [](long thr) { return (unsigned)((thr + 255) / 256); };

  auto convbn = [&](const _Float16* in, int H, int W, int CinAct, int CinW,
                    const float* w, int Cout, _Float16* outH, float* outN) {
    int P = 2 * H * W;
    long n = 9L * Cout * CinAct;
    wt_conv3_k<<<gb(n), 256, 0, stream>>>(w, wtbuf, Cout, CinW, CinAct);
    long thr = (long)(P / 16) * (Cout / 16) * 32;
    conv3x3_wmma_k<<<gb(thr), 256, 0, stream>>>(in, wtbuf, gbuf, 2, H, W, CinAct, Cout);
    bn_partial_k<<<BNCH, 256, 0, stream>>>(gbuf, bnpart, P, Cout);
    bn_finalize_k<<<gb(Cout), 256, 0, stream>>>(bnpart, bnstats, P, Cout);
    long ne = (long)P * Cout;
    bn_apply_k<<<gb(ne), 256, 0, stream>>>(gbuf, bnstats, outH, outN, P, Cout, H * W);
  };
  auto pool = [&](const _Float16* in, int H, int W, int C, _Float16* o) {
    long n = 2L * (H / 2) * (W / 2) * C;
    maxpool2_k<<<gb(n), 256, 0, stream>>>(in, o, 2, H, W, C);
  };

  // --- UNet conv trunk ---
  nchw_pad_k<<<gb(2L * 65536 * 32), 256, 0, stream>>>(F(0), pa, 2, 3, 65536, 32);
  convbn(pa, 256, 256, 32, 3, F(1), 64, pb, nullptr);
  convbn(pb, 256, 256, 64, 64, F(2), 64, pa, ox1);
  pool(pa, 256, 256, 64, pb);
  convbn(pb, 128, 128, 64, 64, F(3), 128, pa, nullptr);
  convbn(pa, 128, 128, 128, 128, F(4), 128, x2f, ox2);
  pool(x2f, 128, 128, 128, pa);
  convbn(pa, 64, 64, 128, 128, F(5), 256, pb, nullptr);
  convbn(pb, 64, 64, 256, 256, F(6), 256, x3f, ox3);
  pool(x3f, 64, 64, 256, pa);
  convbn(pa, 32, 32, 256, 256, F(7), 512, pb, nullptr);
  convbn(pb, 32, 32, 512, 512, F(8), 512, x4f, ox4);
  pool(x4f, 32, 32, 512, pa);
  convbn(pa, 16, 16, 512, 512, F(9), 1024, pb, nullptr);
  convbn(pb, 16, 16, 1024, 1024, F(10), 1024, x5f, nullptr);
  wof = ck;

  // --- DEAM parameter table (d_in flattening order) ---
  struct DeamP {
    const float *proj_w[4], *proj_b[4], *lvl, *off_w, *off_b, *aw_w, *aw_b,
        *vw, *vb, *ow, *ob, *ln1g, *ln1b, *fw1, *fb1, *fw2, *fb2, *ln2g, *ln2b;
  };
  DeamP dps[4];
  int Ls[4] = {1, 2, 3, 4};
  int ci = 13;
  for (int dI = 0; dI < 4; ++dI) {
    DeamP& p = dps[dI];
    int L = Ls[dI];
    for (int l = 0; l < L; ++l) p.proj_w[l] = F(ci++);
    for (int l = 0; l < L; ++l) p.proj_b[l] = F(ci++);
    p.lvl = F(ci++); p.off_w = F(ci++); p.off_b = F(ci++);
    p.aw_w = F(ci++); p.aw_b = F(ci++);
    p.vw = F(ci++); p.vb = F(ci++); p.ow = F(ci++); p.ob = F(ci++);
    p.ln1g = F(ci++); p.ln1b = F(ci++);
    p.fw1 = F(ci++); p.fb1 = F(ci++); p.fw2 = F(ci++); p.fb2 = F(ci++);
    p.ln2g = F(ci++); p.ln2b = F(ci++);
  }

  struct LvlIn { const _Float16* a; int C, h, w; };

  auto gemmF = [&](const float* A, const _Float16* Wt, const float* b0, const float* b1,
                   float* o, int M, int K, int N, int act) {
    long thr = (long)(M / 16) * (N / 16) * 32;
    gemm_wmma_k<float><<<gb(thr), 256, 0, stream>>>(A, Wt, b0, b1, o, M, K, N,
                                                    M, 0, 0, act);
  };

  auto runDeam = [&](const LvlIn* lv, int L, const DeamP& pp, _Float16* tout) {
    int st[4] = {0, 0, 0, 0}, lh[4] = {1, 1, 1, 1}, lw[4] = {1, 1, 1, 1};
    int Lq = 0;
    for (int l = 0; l < L; ++l) {
      st[l] = Lq; lh[l] = lv[l].h; lw[l] = lv[l].w;
      Lq += lv[l].h * lv[l].w;
    }
    int M = 2 * Lq;
    size_t ck2 = wof;
    float* src = af((size_t)M * 128);
    float* val = af((size_t)M * 128);
    float* samp = af((size_t)M * 128);
    float* src2 = af((size_t)M * 128);
    float* awb = af((size_t)M * 32 * L);
    float* big = af((size_t)M * 512);               // off (<=64L<=256) then ffn hidden
    _Float16* wl = ah((size_t)128 * 512);

    // level projections + level_embed -> src tokens
    for (int l = 0; l < L; ++l) {
      long n = (long)128 * lv[l].C;
      wt_oi_k<<<gb(n), 256, 0, stream>>>(pp.proj_w[l], wl, n);
      int Ml = 2 * lv[l].h * lv[l].w;
      long thr = (long)(Ml / 16) * (128 / 16) * 32;
      gemm_wmma_k<_Float16><<<gb(thr), 256, 0, stream>>>(
          lv[l].a, wl, pp.proj_b[l], pp.lvl + (size_t)l * 128, src,
          Ml, lv[l].C, 128, lv[l].h * lv[l].w, (long)Lq * 128, (long)st[l] * 128, 0);
    }
    // value
    wt_lin_k<<<gb(128L * 128), 256, 0, stream>>>(pp.vw, wl, 128, 128);
    gemmF(src, wl, pp.vb, nullptr, val, M, 128, 128, 0);
    // sampling offsets
    int NO = 64 * L;
    wt_lin_k<<<gb(128L * NO), 256, 0, stream>>>(pp.off_w, wl, 128, NO);
    gemmF(src, wl, pp.off_b, nullptr, big, M, 128, NO, 0);
    // attention weights + softmax
    int NA = 32 * L;
    wt_lin_k<<<gb(128L * NA), 256, 0, stream>>>(pp.aw_w, wl, 128, NA);
    gemmF(src, wl, pp.aw_b, nullptr, awb, M, 128, NA, 0);
    softmax_k<<<gb((long)M * 8), 256, 0, stream>>>(awb, (long)M * 8, 4 * L);
    // deformable sampling
    msda_sample_k<<<gb((long)M * 128), 256, 0, stream>>>(
        val, big, awb, samp, Lq, L,
        lh[0], lw[0], st[0], lh[1], lw[1], st[1],
        lh[2], lw[2], st[2], lh[3], lw[3], st[3]);
    // output projection + residual LN
    wt_lin_k<<<gb(128L * 128), 256, 0, stream>>>(pp.ow, wl, 128, 128);
    gemmF(samp, wl, pp.ob, nullptr, src2, M, 128, 128, 0);
    add_ln_k<<<M, 32, 0, stream>>>(src, src2, pp.ln1g, pp.ln1b, M);
    // FFN (gelu) + residual LN
    wt_lin_k<<<gb(128L * 512), 256, 0, stream>>>(pp.fw1, wl, 128, 512);
    gemmF(src, wl, pp.fb1, nullptr, big, M, 128, 512, 2);
    wt_lin_k<<<gb(512L * 128), 256, 0, stream>>>(pp.fw2, wl, 512, 128);
    gemmF(big, wl, pp.fb2, nullptr, src2, M, 512, 128, 0);
    add_ln_k<<<M, 32, 0, stream>>>(src, src2, pp.ln2g, pp.ln2b, M);
    // extract last-level tokens back to f16 NHWC
    int hwl = lv[L - 1].h * lv[L - 1].w;
    tok_to_f16_k<<<gb(2L * hwl * 128), 256, 0, stream>>>(src, tout, Lq, st[L - 1], hwl);
    wof = ck2;
  };

  LvlIn l1[1] = {{x2f, 128, 128, 128}};
  runDeam(l1, 1, dps[0], t1f);
  LvlIn l2[2] = {{x2f, 128, 128, 128}, {x3f, 256, 64, 64}};
  runDeam(l2, 2, dps[1], t2f);
  LvlIn l3[3] = {{x2f, 128, 128, 128}, {x3f, 256, 64, 64}, {x4f, 512, 32, 32}};
  runDeam(l3, 3, dps[2], t3f);
  LvlIn l4[4] = {{t1f, 128, 128, 128}, {t2f, 128, 64, 64},
                 {t3f, 128, 32, 32}, {x5f, 1024, 16, 16}};
  runDeam(l4, 4, dps[3], t4f);

  // --- adj: 1x1 conv 128->1024 (+bias) -> BN -> ReLU -> b output ---
  wof = ck;
  _Float16* wl2 = ah((size_t)1024 * 128);
  float* g2 = af((size_t)512 * 1024);
  float* bp2 = af((size_t)BNCH * 1024 * 2);
  float* bs2 = af((size_t)1024 * 2);
  wt_oi_k<<<gb(1024L * 128), 256, 0, stream>>>(F(11), wl2, 1024L * 128);
  {
    long thr = (long)(512 / 16) * (1024 / 16) * 32;
    gemm_wmma_k<_Float16><<<gb(thr), 256, 0, stream>>>(
        t4f, wl2, F(12), nullptr, g2, 512, 128, 1024, 512, 0, 0, 0);
  }
  bn_partial_k<<<BNCH, 256, 0, stream>>>(g2, bp2, 512, 1024);
  bn_finalize_k<<<gb(1024), 256, 0, stream>>>(bp2, bs2, 512, 1024);
  bn_apply_k<<<gb(512L * 1024), 256, 0, stream>>>(g2, bs2, nullptr, oxb, 512, 1024, 256);
}
